// GCN_83811991814571
// MI455X (gfx1250) — matrix operand order repaired and checked by reading, not compiled, not run on hardware
//
#include <hip/hip_runtime.h>

// ---------------------------------------------------------------------------
// GCN: out = tanh(spmm @ W2) ∘ tanh(spmm @ W1) ∘ tanh(spmm @ W0)
// N=100000 nodes, E=3.2M edges, H=128, C=40.
// SpMM is L2-gather bound (node matrix fits in 192MB L2); GEMMs use
// v_wmma_f32_16x16x32_f16 with f32 accumulation.
// ---------------------------------------------------------------------------

typedef _Float16 v16h __attribute__((ext_vector_type(16)));
typedef _Float16 h4   __attribute__((ext_vector_type(4)));
typedef float    v8f  __attribute__((ext_vector_type(8)));

#define H_DIM 128

// ---------------------------------------------------------------------------
// Zero-fill f32 buffer (SpMM accumulator) with float4 stores.
// ---------------------------------------------------------------------------
__global__ void zero_f32_kernel(float4* __restrict__ p, int n4) {
    int i = blockIdx.x * blockDim.x + threadIdx.x;
    if (i < n4) p[i] = make_float4(0.f, 0.f, 0.f, 0.f);
}

// ---------------------------------------------------------------------------
// Row-gather helpers: 4 columns per lane.
// ---------------------------------------------------------------------------
__device__ __forceinline__ float4 loadRow4(const float* p) {
    return *reinterpret_cast<const float4*>(p);          // global_load_b128
}
__device__ __forceinline__ float4 loadRow4(const _Float16* p) {
    h4 v = *reinterpret_cast<const h4*>(p);              // global_load_b64
    return make_float4((float)v[0], (float)v[1], (float)v[2], (float)v[3]);
}

// ---------------------------------------------------------------------------
// SpMM: outS[d] += w_e * hin[s] over edges, edge_dst sorted ascending.
// Block = (32, 8): one wave per edge-slice; 32 lanes own 4 columns each
// (128 cols). Register accumulation across runs of equal dst, flushed with
// f32 global atomics at run boundaries (~1 flush per 32 edges).
// Software-prefetch the gathered row PF_DIST edges ahead (edge index stream
// is sequential and cheap to peek; the row gather is random in L2).
// ---------------------------------------------------------------------------
template <typename T>
__global__ void spmm_kernel(const T* __restrict__ hin,
                            const int* __restrict__ esrc,
                            const int* __restrict__ edst,
                            const float* __restrict__ ew,
                            float* __restrict__ outS,
                            int nEdges, int edgesPerSlice) {
    constexpr int PF_DIST = 8;
    const int c     = 4 * threadIdx.x;                 // 4-column group
    const int slice = blockIdx.x * blockDim.y + threadIdx.y;
    long e0 = (long)slice * edgesPerSlice;
    if (e0 >= nEdges) return;
    long e1 = e0 + edgesPerSlice;
    if (e1 > nEdges) e1 = nEdges;

    float4 acc = make_float4(0.f, 0.f, 0.f, 0.f);
    int prev = -1;
    for (long e = e0; e < e1; ++e) {
        // Prefetch the row gathered PF_DIST edges ahead into the WGP cache.
        long ef = e + PF_DIST;
        if (ef >= e1) ef = e1 - 1;
        const int sf = esrc[ef];
        __builtin_prefetch(hin + (size_t)sf * H_DIM + c, 0, 0);

        const int   s = esrc[e];
        const int   d = edst[e];
        const float w = ew[e];
        if (d != prev) {
            if (prev >= 0) {
                float* o = &outS[(size_t)prev * H_DIM + c];
                atomicAdd(o + 0, acc.x);
                atomicAdd(o + 1, acc.y);
                atomicAdd(o + 2, acc.z);
                atomicAdd(o + 3, acc.w);
            }
            acc = make_float4(0.f, 0.f, 0.f, 0.f);
            prev = d;
        }
        const float4 r = loadRow4(hin + (size_t)s * H_DIM + c);
        acc.x = fmaf(w, r.x, acc.x);
        acc.y = fmaf(w, r.y, acc.y);
        acc.z = fmaf(w, r.z, acc.z);
        acc.w = fmaf(w, r.w, acc.w);
    }
    if (prev >= 0) {
        float* o = &outS[(size_t)prev * H_DIM + c];
        atomicAdd(o + 0, acc.x);
        atomicAdd(o + 1, acc.y);
        atomicAdd(o + 2, acc.z);
        atomicAdd(o + 3, acc.w);
    }
}

// ---------------------------------------------------------------------------
// Y = tanh(S @ W), S: [nrows x 128] f32, W: [128 x houtReal] f32.
// HOUT_PAD = houtReal padded to multiple of 16. Block = 256 threads = 8 waves;
// each wave owns a 16-row strip and all HOUT_PAD/16 column tiles.
// All four 16x32 A fragments are preloaded/converted up front (16 independent
// global_load_b128 in flight), then the K loop is purely LDS-B + WMMA.
// ---------------------------------------------------------------------------
template <int HOUT_PAD, typename OutT>
__global__ void gemm_tanh_kernel(const float* __restrict__ S,
                                 const float* __restrict__ W,
                                 OutT* __restrict__ out,
                                 int nrows, int houtReal) {
    constexpr int NCT = HOUT_PAD / 16;
    __shared__ _Float16 Wt[HOUT_PAD * H_DIM];   // Wt[n*128 + k] = (f16) W[k][n]

    const int tid = threadIdx.x;

    // Stage W -> LDS (transposed, f16, zero-padded columns).
    for (int i = tid; i < HOUT_PAD * H_DIM; i += blockDim.x) {
        const int n = i / H_DIM;
        const int k = i % H_DIM;
        float v = (n < houtReal) ? W[(size_t)k * houtReal + n] : 0.f;
        Wt[i] = (_Float16)v;
    }
    __syncthreads();

    const int wave = tid >> 5;          // 0..7
    const int lane = tid & 31;
    const int lm   = lane & 15;
    const int row0 = blockIdx.x * 128 + wave * 16;

    // --- Preload all A fragments (4 K-slices of 16x32 f16, contiguous-K layout) ---
    // lane<16: M=lm, K = k0+0..15; lane>=16: M=lm, K = k0+16..31
    const int m  = row0 + lm;
    const int mc = (m < nrows) ? m : 0;              // clamp; stores are masked
    const float* __restrict__ srow = S + (size_t)mc * H_DIM;

    v16h afrag[4];
#pragma unroll
    for (int ks = 0; ks < 4; ++ks) {
        const int kb1 = ks * 32 + ((lane < 16) ? 0 : 16);
        const int kb2 = kb1 + 8;
        const float4* pa1 = reinterpret_cast<const float4*>(srow + kb1);
        const float4* pa2 = reinterpret_cast<const float4*>(srow + kb2);
        float4 f0 = pa1[0], f1 = pa1[1];
        float4 f2 = pa2[0], f3 = pa2[1];
        v16h a;
        a[0]  = (_Float16)f0.x; a[1]  = (_Float16)f0.y; a[2]  = (_Float16)f0.z; a[3]  = (_Float16)f0.w;
        a[4]  = (_Float16)f1.x; a[5]  = (_Float16)f1.y; a[6]  = (_Float16)f1.z; a[7]  = (_Float16)f1.w;
        a[8]  = (_Float16)f2.x; a[9]  = (_Float16)f2.y; a[10] = (_Float16)f2.z; a[11] = (_Float16)f2.w;
        a[12] = (_Float16)f3.x; a[13] = (_Float16)f3.y; a[14] = (_Float16)f3.z; a[15] = (_Float16)f3.w;
        afrag[ks] = a;
    }

    v8f acc[NCT];
#pragma unroll
    for (int ct = 0; ct < NCT; ++ct) acc[ct] = {};

    // --- K loop: LDS B fragments + WMMA only ---
    // lane<16: N=ct*16+lm, K = k0+0..15; lane>=16: same N, K = k0+16..31
#pragma unroll
    for (int ks = 0; ks < 4; ++ks) {
        const int kh = ks * 32 + ((lane < 16) ? 0 : 16);
#pragma unroll
        for (int ct = 0; ct < NCT; ++ct) {
            const int n = ct * 16 + lm;
            const v16h b = *reinterpret_cast<const v16h*>(&Wt[n * H_DIM + kh]);
            acc[ct] = __builtin_amdgcn_wmma_f32_16x16x32_f16(
                /*neg_a=*/false, afrag[ks], /*neg_b=*/false, b,
                /*c_mod=*/(short)0, acc[ct],
                /*reuse_a=*/false, /*reuse_b=*/false);
        }
    }

    // --- Epilogue: tanh + masked store ---
    // C/D layout: VGPR r holds M = r (lanes 0-15) / M = r+8 (lanes 16-31); N = lane&15.
    const int mofs = (lane < 16) ? 0 : 8;
#pragma unroll
    for (int ct = 0; ct < NCT; ++ct) {
        const int n = ct * 16 + lm;
        if (n >= houtReal) continue;
#pragma unroll
        for (int r = 0; r < 8; ++r) {
            const int mrow = row0 + r + mofs;
            if (mrow < nrows) {
                float t = tanhf(acc[ct][r]);
                out[(size_t)mrow * houtReal + n] = (OutT)t;
            }
        }
    }
}

// ---------------------------------------------------------------------------
// Host-side launch
// ---------------------------------------------------------------------------
extern "C" void kernel_launch(void* const* d_in, const int* in_sizes, int n_in,
                              void* d_out, int out_size, void* d_ws, size_t ws_size,
                              hipStream_t stream) {
    const float* x    = (const float*)d_in[0];   // [N, 128]
    const int*   esrc = (const int*)  d_in[1];   // [E]
    const int*   edst = (const int*)  d_in[2];   // [E] sorted
    const float* ew   = (const float*)d_in[3];   // [E]
    const float* W0   = (const float*)d_in[4];   // [128, 128]
    const float* W1   = (const float*)d_in[5];   // [128, 128]
    const float* W2   = (const float*)d_in[6];   // [128, C]
    float*       out  = (float*)d_out;           // [N, C]

    const int N = in_sizes[0] / H_DIM;
    const int E = in_sizes[1];
    const int C = in_sizes[6] / H_DIM;           // 40

    // Workspace: S (f32 accumulator, N*128) then h (f16 activations, N*128).
    float*     S    = (float*)d_ws;
    _Float16*  Hbuf = (_Float16*)((char*)d_ws + (size_t)N * H_DIM * sizeof(float));

    const int n4       = (N * H_DIM) / 4;
    const int zeroGrid = (n4 + 255) / 256;

    const int EDGES_PER_SLICE = 128;
    const int slices   = (E + EDGES_PER_SLICE - 1) / EDGES_PER_SLICE;
    const dim3 spmmBlk(32, 8);
    const int  spmmGrid = (slices + 7) / 8;

    const int gemmGrid = (N + 127) / 128;

    // ---- Layer 1: S = A @ x ; h = tanh(S @ W0) ----
    zero_f32_kernel<<<zeroGrid, 256, 0, stream>>>((float4*)S, n4);
    spmm_kernel<float><<<spmmGrid, spmmBlk, 0, stream>>>(x, esrc, edst, ew, S, E, EDGES_PER_SLICE);
    gemm_tanh_kernel<128, _Float16><<<gemmGrid, 256, 0, stream>>>(S, W0, Hbuf, N, H_DIM);

    // ---- Layer 2 ----
    zero_f32_kernel<<<zeroGrid, 256, 0, stream>>>((float4*)S, n4);
    spmm_kernel<_Float16><<<spmmGrid, spmmBlk, 0, stream>>>(Hbuf, esrc, edst, ew, S, E, EDGES_PER_SLICE);
    gemm_tanh_kernel<128, _Float16><<<gemmGrid, 256, 0, stream>>>(S, W1, Hbuf, N, H_DIM);

    // ---- Output layer: C=40, padded to 48 (3 column tiles) ----
    zero_f32_kernel<<<zeroGrid, 256, 0, stream>>>((float4*)S, n4);
    spmm_kernel<_Float16><<<spmmGrid, spmmBlk, 0, stream>>>(Hbuf, esrc, edst, ew, S, E, EDGES_PER_SLICE);
    gemm_tanh_kernel<48, float><<<gemmGrid, 256, 0, stream>>>(S, W2, out, N, C);
}
